// VectorQuantizer_12876311953427
// MI455X (gfx1250) — matrix operand order repaired and checked by reading, not compiled, not run on hardware
//
#include <hip/hip_runtime.h>

// ---------------------------------------------------------------------------
// VQ-VAE vector quantizer for MI455X (gfx1250, wave32, WMMA).
//   inputs:  d_in[0] = inputs  [32,4096,64] f32   (N=131072 rows, D=64)
//            d_in[1] = embedding [1024,64] f32    (K=1024 codes)
//   outputs (concatenated f32): [0]=loss, [1..N*D]=quantized_st, then N indices
//
// argmin_k ||x - e_k||^2 = argmax_k (x.e_k - 0.5*||e_k||^2); the ||x||^2 term
// is invariant, and the -0.5*||e_k||^2 term is folded into the WMMA C operand,
// so the post-WMMA work is a pure compare/select argmax.
// ---------------------------------------------------------------------------

#define K_CODES        1024
#define DIM            64
#define N_ROWS         131072          // 32*4096
#define TILE_K         128             // codes cached in LDS per tile
#define ROWS_PER_BLOCK 128             // 8 waves * 16 rows
#define THREADS        256
#define LDS_STRIDE     72              // halves per code row in LDS (bank padding)

typedef __attribute__((ext_vector_type(16))) __bf16 v16bf;
typedef __attribute__((ext_vector_type(8)))  __bf16 v8bf;
typedef __attribute__((ext_vector_type(8)))  float  v8f;

// ------------- prep: -0.5*||e||^2 + bf16 codebook + zero loss ---------------
__global__ void vq_prep(const float* __restrict__ emb,
                        __bf16* __restrict__ ebf,
                        float* __restrict__ ehalfneg,
                        float* __restrict__ loss_out) {
  int k = blockIdx.x * blockDim.x + threadIdx.x;
  if (k == 0) loss_out[0] = 0.0f;
  if (k < K_CODES) {
    const float* r = emb + (size_t)k * DIM;
    float s = 0.0f;
#pragma unroll
    for (int d = 0; d < DIM; ++d) {
      float v = r[d];
      s = fmaf(v, v, s);
      ebf[(size_t)k * DIM + d] = (__bf16)v;
    }
    ehalfneg[k] = -0.5f * s;           // WMMA C seed: score = x.e - 0.5||e||^2
  }
}

// ---------------- main: scores via WMMA, argmax, gather, loss ---------------
__launch_bounds__(THREADS)
__global__ void vq_main(const float*  __restrict__ x,
                        const float*  __restrict__ emb,
                        const __bf16* __restrict__ ebf,
                        const float*  __restrict__ ehalfneg,
                        float* __restrict__ out_loss,
                        float* __restrict__ out_q,
                        float* __restrict__ out_idx) {
  __shared__ __bf16 se[TILE_K * LDS_STRIDE];   // bf16 code tile (padded rows)
  __shared__ float  sen[TILE_K];               // -0.5*||e||^2 tile
  __shared__ int    scode[ROWS_PER_BLOCK];     // winning code per row
  __shared__ float  slsum[THREADS / 32];       // per-wave loss partials

  const int t    = threadIdx.x;
  const int l    = t & 31;
  const int w    = t >> 5;          // wave id 0..7
  const int lo16 = l & 15;
  const int hi   = l >> 4;          // lane half 0/1
  const int rowBase = blockIdx.x * ROWS_PER_BLOCK;
  const int myRow   = rowBase + w * 16 + lo16;   // A-matrix row for this lane

  // ---- build A fragments (16x32 bf16, ISA 7.12.2 layout) ----
  // lane covers K segments: [8*hi,+8) [16+8*hi,+8) [32+8*hi,+8) [48+8*hi,+8)
  const float* xr = x + (size_t)myRow * DIM;
  const int o = hi * 8;
  float seg[4][8];
#pragma unroll
  for (int s = 0; s < 4; ++s) {
    const float4 p0 = *(const float4*)(xr + s * 16 + o);
    const float4 p1 = *(const float4*)(xr + s * 16 + o + 4);
    seg[s][0] = p0.x; seg[s][1] = p0.y; seg[s][2] = p0.z; seg[s][3] = p0.w;
    seg[s][4] = p1.x; seg[s][5] = p1.y; seg[s][6] = p1.z; seg[s][7] = p1.w;
  }
  v16bf A0, A1;
#pragma unroll
  for (int h = 0; h < 8; ++h) {
    A0[h]     = (__bf16)seg[0][h];   // K 0..31 fragment
    A0[h + 8] = (__bf16)seg[1][h];
    A1[h]     = (__bf16)seg[2][h];   // K 32..63 fragment
    A1[h + 8] = (__bf16)seg[3][h];
  }

  float bs[8];                        // best score (maximize)
  int   bi[8];                        // best index
#pragma unroll
  for (int r = 0; r < 8; ++r) { bs[r] = -3.4e38f; bi[r] = 0; }

  // ---- sweep the codebook in LDS tiles ----
  const int rr = t >> 1;              // code row handled by this thread
  const int hs = (t & 1) * 32;        // half-row offset (halves)
  for (int kt = 0; kt < K_CODES; kt += TILE_K) {
    {
      const uint4* s4 = (const uint4*)(ebf + (size_t)(kt + rr) * DIM + hs);
      uint4* d4 = (uint4*)(se + rr * LDS_STRIDE + hs);
      d4[0] = s4[0]; d4[1] = s4[1]; d4[2] = s4[2]; d4[3] = s4[3];
      if (t < TILE_K) sen[t] = ehalfneg[kt + t];
      if (kt + TILE_K < K_CODES)      // overlap next tile's HBM fetch
        __builtin_prefetch(ebf + (size_t)(kt + TILE_K + rr) * DIM + hs, 0, 3);
    }
    __syncthreads();

#pragma unroll
    for (int ch = 0; ch < TILE_K / 16; ++ch) {
      const int col = ch * 16 + lo16;                   // code (B column) for lane
      const float c0 = sen[col];                        // -0.5*||e_col||^2
      const __bf16* bp = se + col * LDS_STRIDE + 8 * hi;
      const v8bf b0lo = *(const v8bf*)(bp);             // K [8*hi,+8) [16+8*hi,+8) of 0..31
      const v8bf b0hi = *(const v8bf*)(bp + 16);
      const v8bf b1lo = *(const v8bf*)(bp + 32);        // same for K 32..63
      const v8bf b1hi = *(const v8bf*)(bp + 48);
      const v16bf B0 = __builtin_shufflevector(b0lo, b0hi,
          0,1,2,3,4,5,6,7,8,9,10,11,12,13,14,15);
      const v16bf B1 = __builtin_shufflevector(b1lo, b1hi,
          0,1,2,3,4,5,6,7,8,9,10,11,12,13,14,15);

      v8f acc = {c0, c0, c0, c0, c0, c0, c0, c0};       // fold -0.5||e||^2 in C
      acc = __builtin_amdgcn_wmma_f32_16x16x32_bf16(false, A0, false, B0,
                                                    (short)0, acc, false, false);
      acc = __builtin_amdgcn_wmma_f32_16x16x32_bf16(false, A1, false, B1,
                                                    (short)0, acc, false, false);

      const int cg = kt + col;
#pragma unroll
      for (int r = 0; r < 8; ++r) {
        const bool p = acc[r] > bs[r];                  // strict: keep 1st max
        bs[r] = p ? acc[r] : bs[r];
        bi[r] = p ? cg     : bi[r];
      }
    }
    __syncthreads();
  }

  // ---- argmax across the 16 lanes holding each C row ----
#pragma unroll
  for (int m = 1; m <= 8; m <<= 1) {
#pragma unroll
    for (int r = 0; r < 8; ++r) {
      const float os = __shfl_xor(bs[r], m);
      const int   oi = __shfl_xor(bi[r], m);
      if (os > bs[r] || (os == bs[r] && oi < bi[r])) { bs[r] = os; bi[r] = oi; }
    }
  }
  if (lo16 == 0) {
#pragma unroll
    for (int r = 0; r < 8; ++r) scode[w * 16 + r + 8 * hi] = bi[r];
  }
  __syncthreads();

  // ---- phase 2: gather f32 code rows -> quantized output, indices, loss ----
  float lacc = 0.0f;
#pragma unroll 4
  for (int it = 0; it < (ROWS_PER_BLOCK * DIM) / THREADS; ++it) {
    const int g  = it * THREADS + t;
    const int m  = g >> 6;
    const int dd = g & 63;
    const int c  = scode[m];
    const float e  = emb[(size_t)c * DIM + dd];
    const float xv = x[(size_t)(rowBase + m) * DIM + dd];
    out_q[(size_t)(rowBase + m) * DIM + dd] = e;
    const float df = e - xv;
    lacc = fmaf(df, df, lacc);
    if (dd == 0) out_idx[rowBase + m] = (float)c;
  }
#pragma unroll
  for (int m = 1; m < 32; m <<= 1) lacc += __shfl_xor(lacc, m);
  if (l == 0) slsum[w] = lacc;
  __syncthreads();
  if (t == 0) {
    float s = 0.0f;
#pragma unroll
    for (int i = 0; i < THREADS / 32; ++i) s += slsum[i];
    // loss = q_latent + 0.25*e_latent = 1.25 * mean((q-x)^2)
    atomicAdd(out_loss, s * (1.25f / 8388608.0f));   // N_ROWS*DIM = 8388608
  }
}

// ---------------------------------------------------------------------------
extern "C" void kernel_launch(void* const* d_in, const int* in_sizes, int n_in,
                              void* d_out, int out_size, void* d_ws, size_t ws_size,
                              hipStream_t stream) {
  (void)in_sizes; (void)n_in; (void)out_size; (void)ws_size;
  const float* x   = (const float*)d_in[0];
  const float* emb = (const float*)d_in[1];

  float* out      = (float*)d_out;
  float* out_loss = out;
  float* out_q    = out + 1;
  float* out_idx  = out + 1 + (size_t)N_ROWS * DIM;

  __bf16* ebf  = (__bf16*)d_ws;                                  // 128 KB
  float*  ehn  = (float*)((char*)d_ws + (size_t)K_CODES * DIM * sizeof(__bf16)); // 4 KB

  vq_prep<<<(K_CODES + 255) / 256, 256, 0, stream>>>(emb, ebf, ehn, out_loss);
  vq_main<<<N_ROWS / ROWS_PER_BLOCK, THREADS, 0, stream>>>(
      x, emb, ebf, ehn, out_loss, out_q, out_idx);
}
